// SpMiddleResNetFHD_83279415869714
// MI455X (gfx1250) — matrix operand order repaired and checked by reading, not compiled, not run on hardware
//
#include <hip/hip_runtime.h>
#include <math.h>

typedef __attribute__((ext_vector_type(16))) _Float16 v16h;
typedef __attribute__((ext_vector_type(8)))  float    v8f;
typedef __attribute__((ext_vector_type(4)))  int      v4i;
typedef __attribute__((ext_vector_type(8)))  int      v8i;

__device__ __forceinline__ v8f wmma_step(v4i a0, v4i a1, v4i b0, v4i b1, v8f acc) {
  v8i ai, bi;
  ai[0]=a0[0]; ai[1]=a0[1]; ai[2]=a0[2]; ai[3]=a0[3];
  ai[4]=a1[0]; ai[5]=a1[1]; ai[6]=a1[2]; ai[7]=a1[3];
  bi[0]=b0[0]; bi[1]=b0[1]; bi[2]=b0[2]; bi[3]=b0[3];
  bi[4]=b1[0]; bi[5]=b1[1]; bi[6]=b1[2]; bi[7]=b1[3];
  v16h a = __builtin_bit_cast(v16h, ai);
  v16h b = __builtin_bit_cast(v16h, bi);
  return __builtin_amdgcn_wmma_f32_16x16x32_f16(false, a, false, b, (short)0,
                                                acc, false, false);
}

// ---------------------------------------------------------------------------
// Implicit-GEMM 3-D conv via v_wmma_f32_16x16x32_f16.
//   act: halo-padded channels-last f16, [(D+2)(H+2)(W+2)][Cin], halo = 0
//   wt : [Cout][K32] f16, k' = tap*Cin + c, zero-padded to K32
//   out: [Cout][Mout] f32
// One wave = 32(M) x 16(N); B fragments shared by both accumulators.
// No bounds checks (halo), no LDS (incremental scalar tap offsets),
// branch-free inner loop: 6x global_load_b128 + 2x WMMA per step.
// ---------------------------------------------------------------------------
template <int LC>
__global__ __launch_bounds__(256) void conv3d_wmma(
    const _Float16* __restrict__ act, const _Float16* __restrict__ wt,
    float* __restrict__ out,
    int taps, int Hp, int Wp,           // padded input H+2, W+2
    int Dout, int Hout, int Wout,
    int sd, int sh, int sw, int pd, int ph, int pw,
    int KH, int KW)
{
  constexpr int Cin = 1 << LC;
  const int lane  = threadIdx.x & 31;
  const int wv    = threadIdx.x >> 5;
  const int Mtot  = Dout * Hout * Wout;
  const int mtile = (blockIdx.x * 8 + wv) * 32;
  const int ntile = blockIdx.y * 16;
  if (mtile >= Mtot) return;            // uniform per wave -> EXEC all-1

  const int half = lane >> 4;
  const int l15  = lane & 15;
  const int HW   = Hout * Wout;

  auto rowbase = [&](int m) -> const _Float16* {
    if (m >= Mtot) m = Mtot - 1;        // clamped rows discarded at store
    int oz = m / HW; int r = m - oz * HW;
    int oy = r / Wout; int ox = r - oy * Wout;
    int bz = oz * sd - pd + 1, by = oy * sh - ph + 1, bx = ox * sw - pw + 1;
    int vlin = (bz * Hp + by) * Wp + bx;
    return act + (((size_t)vlin) << LC) + half * 8;
  };
  const _Float16* aB0 = rowbase(mtile + l15);
  const _Float16* aB1 = rowbase(mtile + 16 + l15);

  const int K32 = (Cin * taps + 31) & ~31;
  const int n = ntile + l15;
  const _Float16* wrow = wt + (size_t)n * K32 + half * 8;

  // incremental tap offset (scalar bookkeeping, no LDS, no division)
  int off = 0, dx = 0, dy = 0;
  const int stepY = Wp - KW, stepZ = (Hp - KH) * Wp;
  auto adv = [&]() {
    off += 1;
    if (++dx == KW) { dx = 0; off += stepY;
      if (++dy == KH) { dy = 0; off += stepZ; } }
  };

  v8f acc0 = {}, acc1 = {};
  if constexpr (LC == 4) {
    // Cin=16: one WMMA covers a tap pair (run0 -> even tap, run1 -> odd)
    const int np = (taps + 1) >> 1;
    for (int tp = 0; tp < np; ++tp) {
      const size_t o0 = (size_t)off << 4;
      adv();
      const bool has1 = (2 * tp + 1) < taps;      // phantom tap: zero weights
      const size_t o1 = has1 ? ((size_t)off << 4) : o0;
      adv();
      v4i a00 = *(const v4i*)(aB0 + o0);
      v4i a01 = *(const v4i*)(aB0 + o1);
      v4i a10 = *(const v4i*)(aB1 + o0);
      v4i a11 = *(const v4i*)(aB1 + o1);
      const _Float16* pw = wrow + tp * 32;
      v4i b0 = *(const v4i*)pw;
      v4i b1 = *(const v4i*)(pw + 16);
      acc0 = wmma_step(a00, a01, b0, b1, acc0);
      acc1 = wmma_step(a10, a11, b0, b1, acc1);
    }
  } else {
    constexpr int CPT = Cin >> 5;       // 32-wide K chunks per tap
    for (int t = 0; t < taps; ++t) {
      const _Float16* pa0 = aB0 + ((size_t)off << LC);
      const _Float16* pa1 = aB1 + ((size_t)off << LC);
      const _Float16* pw  = wrow + t * Cin;
      adv();
#pragma unroll
      for (int cc = 0; cc < CPT; ++cc) {
        v4i a00 = *(const v4i*)(pa0 + cc * 32);
        v4i a01 = *(const v4i*)(pa0 + cc * 32 + 16);
        v4i a10 = *(const v4i*)(pa1 + cc * 32);
        v4i a11 = *(const v4i*)(pa1 + cc * 32 + 16);
        v4i b0  = *(const v4i*)(pw + cc * 32);
        v4i b1  = *(const v4i*)(pw + cc * 32 + 16);
        acc0 = wmma_step(a00, a01, b0, b1, acc0);
        acc1 = wmma_step(a10, a11, b0, b1, acc1);
      }
    }
  }
  // D store: VGPR v -> M = v + 8*half, N = l15
#pragma unroll
  for (int v = 0; v < 8; ++v) {
    int mr0 = mtile + v + half * 8;
    if (mr0 < Mtot) out[(size_t)n * Mtot + mr0] = acc0[v];
    int mr1 = mtile + 16 + v + half * 8;
    if (mr1 < Mtot) out[(size_t)n * Mtot + mr1] = acc1[v];
  }
}

// Repack fp32 [Cout][Cin][taps] weights -> f16 [Cout][K32], k' = tap*Cin + c.
__global__ void repack_w_k(const float* __restrict__ w, _Float16* __restrict__ wt,
                           int K32, int lc, int taps, int tot) {
  int i = blockIdx.x * blockDim.x + threadIdx.x;
  if (i >= tot) return;
  int n  = i / K32;
  int kk = i - n * K32;
  int tap = kk >> lc;
  int c   = kk & ((1 << lc) - 1);
  int Cin = 1 << lc;
  float v = (tap < taps) ? w[(size_t)n * Cin * taps + (size_t)c * taps + tap] : 0.f;
  wt[i] = (_Float16)v;
}

// grid-stride int4 memset (zero halo-padded f16 buffers)
__global__ void zero_f16_k(v4i* __restrict__ p, int n) {
  v4i z = {};
  for (int i = blockIdx.x * blockDim.x + threadIdx.x; i < n;
       i += gridDim.x * blockDim.x)
    p[i] = z;
}

// f32 [C][M] -> f16 halo-padded [M'][C] (interior only)
__global__ void to_cl16_k(const float* __restrict__ x, _Float16* __restrict__ y,
                          int lcC, int Mv, int H, int W, int Hp, int Wp) {
  int i = blockIdx.x * blockDim.x + threadIdx.x;
  if (i >= (Mv << lcC)) return;
  int v = i >> lcC;
  int c = i & ((1 << lcC) - 1);
  int z = v / (H * W); int r = v - z * (H * W);
  int yy = r / W; int xx = r - yy * W;
  size_t p = ((size_t)((z + 1) * Hp + (yy + 1)) * Wp + (xx + 1)) << lcC;
  y[p + c] = (_Float16)x[(size_t)c * Mv + v];
}

// mask[v] = any channel of features nonzero (features were pre-masked).
__global__ void compute_mask_k(const float* __restrict__ f, float* __restrict__ m,
                               int C, int Mv) {
  int i = blockIdx.x * blockDim.x + threadIdx.x;
  if (i >= Mv) return;
  float s = 0.f;
  for (int c = 0; c < C; ++c) s += fabsf(f[(size_t)c * Mv + i]);
  m[i] = (s > 0.f) ? 1.f : 0.f;
}

__global__ void pool_mask_k(const float* __restrict__ mi, float* __restrict__ mo,
                            int Di, int Hi, int Wi, int sd, int sh, int sw) {
  const int Ho = Hi / sh, Wo = Wi / sw, Do = Di / sd;
  int i = blockIdx.x * blockDim.x + threadIdx.x;
  if (i >= Do * Ho * Wo) return;
  int z = i / (Ho * Wo); int r = i % (Ho * Wo); int y = r / Wo; int x = r % Wo;
  float v = 0.f;
  for (int a = 0; a < sd; ++a)
    for (int b = 0; b < sh; ++b)
      for (int c = 0; c < sw; ++c)
        v = fmaxf(v, mi[((size_t)(z * sd + a) * Hi + y * sh + b) * Wi + x * sw + c]);
  mo[i] = v;
}

__global__ void mask_sum_k(const float* __restrict__ m, float* __restrict__ nout,
                           int Mv) {
  __shared__ float sh[256];
  float s = 0.f;
  for (int i = threadIdx.x; i < Mv; i += 256) s += m[i];
  sh[threadIdx.x] = s;
  __syncthreads();
  for (int st = 128; st > 0; st >>= 1) {
    if ((int)threadIdx.x < st) sh[threadIdx.x] += sh[threadIdx.x + st];
    __syncthreads();
  }
  if (threadIdx.x == 0) nout[0] = fmaxf(sh[0], 1.f);
}

// Per-channel masked sums: stats[2c] = sum(x*m), stats[2c+1] = sum(x^2*m)
__global__ void bn_stats_k(const float* __restrict__ x, const float* __restrict__ m,
                           float* __restrict__ stats, int Mv) {
  __shared__ float s1[256], s2[256];
  const float* xc = x + (size_t)blockIdx.x * Mv;
  float a = 0.f, b = 0.f;
  for (int i = threadIdx.x; i < Mv; i += 256) {
    float mv = m[i];
    float v = xc[i];
    a += v * mv;
    b += v * v * mv;
  }
  s1[threadIdx.x] = a; s2[threadIdx.x] = b;
  __syncthreads();
  for (int st = 128; st > 0; st >>= 1) {
    if ((int)threadIdx.x < st) {
      s1[threadIdx.x] += s1[threadIdx.x + st];
      s2[threadIdx.x] += s2[threadIdx.x + st];
    }
    __syncthreads();
  }
  if (threadIdx.x == 0) {
    stats[2 * blockIdx.x]     = s1[0];
    stats[2 * blockIdx.x + 1] = s2[0];
  }
}

// y = ((x-mean)*rsqrt(var+eps)*g + b)*mask [+res] [relu]
// grid = (ceil(Mv/256), C). Writes f32 [C][M] + optional halo-padded f16.
__global__ void bn_apply_k(const float* __restrict__ x, const float* __restrict__ res,
                           const float* __restrict__ m, const float* __restrict__ g,
                           const float* __restrict__ bb, const float* __restrict__ stats,
                           const float* __restrict__ nptr, float* __restrict__ out,
                           _Float16* __restrict__ out16,
                           int Mv, int lcC, int relu,
                           int H, int W, int Hp, int Wp) {
  int v = blockIdx.x * blockDim.x + threadIdx.x;
  if (v >= Mv) return;
  const int c = blockIdx.y;
  float n = nptr[0];
  float mean = stats[2 * c] / n;
  float var  = stats[2 * c + 1] / n - mean * mean;
  float rs = rsqrtf(var + 1e-5f);
  size_t i = (size_t)c * Mv + v;
  float y = ((x[i] - mean) * rs * g[c] + bb[c]) * m[v];
  if (res) y += res[i];
  if (relu) y = fmaxf(y, 0.f);
  out[i] = y;
  if (out16) {
    int z = v / (H * W); int r = v - z * (H * W);
    int yy = r / W; int xx = r - yy * W;
    size_t p = ((size_t)((z + 1) * Hp + (yy + 1)) * Wp + (xx + 1)) << lcC;
    out16[p + c] = (_Float16)y;
  }
}

// ---------------------------------------------------------------------------
// Host orchestration
// ---------------------------------------------------------------------------
struct CP { const float *w, *g, *b; };
struct BP { CP c1, c2; };

static const int kSortedSz[63] = {
  16,16,16,16,6912,6912,  16,16,16,16,6912,6912,                          // conv1
  32, 32,32,32,32,27648,27648, 32,32,32,32,27648,27648, 32, 13824,        // conv2
  64, 64,64,64,64,110592,110592, 64,64,64,64,110592,110592, 64, 55296,    // conv3
  128, 128,128,128,128,442368,442368, 128,128,128,128,442368,442368, 128, 221184, // conv4
  128,128,49152,                                                          // conv_extra
  16,16,6912                                                              // conv_input
};

extern "C" void kernel_launch(void* const* d_in, const int* in_sizes, int n_in,
                              void* d_out, int out_size, void* d_ws, size_t ws_size,
                              hipStream_t stream) {
  (void)out_size; (void)ws_size;
  const float* FEAT = (const float*)d_in[0];

  const float* L[64];
  L[0] = FEAT;
  bool sortedOrder = true;
  if (n_in >= 64) {
    for (int i = 1; i < 64; ++i) L[i] = (const float*)d_in[i];
    sortedOrder = !(in_sizes[1] > 1000);   // insertion order => conv_input.w first
  } else {
    const float* blob = (const float*)d_in[1];
    size_t off = 0;
    for (int i = 0; i < 63; ++i) { L[i + 1] = blob + off; off += kSortedSz[i]; }
    sortedOrder = true;
  }

  auto blk = [&](int base) -> BP {
    BP r;
    if (!sortedOrder) {  // insertion: w1,g1,b1,w2,g2,b2
      r.c1 = {L[base + 0], L[base + 1], L[base + 2]};
      r.c2 = {L[base + 3], L[base + 4], L[base + 5]};
    } else {             // sorted: b1,b2,g1,g2,w1,w2
      r.c1 = {L[base + 4], L[base + 2], L[base + 0]};
      r.c2 = {L[base + 5], L[base + 3], L[base + 1]};
    }
    return r;
  };

  CP cin_, c2d, c3d, c4d, cex;
  BP b1[2], b2[2], b3[2], b4[2];
  if (!sortedOrder) {
    cin_ = {L[1], L[2], L[3]};
    b1[0] = blk(4);  b1[1] = blk(10);
    c2d = {L[16], L[17], L[18]}; b2[0] = blk(19); b2[1] = blk(25);
    c3d = {L[31], L[32], L[33]}; b3[0] = blk(34); b3[1] = blk(40);
    c4d = {L[46], L[47], L[48]}; b4[0] = blk(49); b4[1] = blk(55);
    cex = {L[61], L[62], L[63]};
  } else {
    b1[0] = blk(1);  b1[1] = blk(7);
    c2d = {L[27], L[26], L[13]}; b2[0] = blk(14); b2[1] = blk(20);
    c3d = {L[42], L[41], L[28]}; b3[0] = blk(29); b3[1] = blk(35);
    c4d = {L[57], L[56], L[43]}; b4[0] = blk(44); b4[1] = blk(50);
    cex = {L[60], L[59], L[58]};
    cin_ = {L[63], L[62], L[61]};
  }

  // ---- workspace layout (float units) --------------------------------------
  float* ws = (float*)d_ws;
  const size_t NB  = 6553600;            // unpadded f32 activations (max)
  const size_t NBP = 7558400;            // halo-padded f16 halves: 18*162*162*16
  float* B0v = ws;
  float* B1v = B0v + NB;
  float* B2v = B1v + NB;
  float* M0 = B2v + NB;                  // 409600
  float* M1 = M0 + 409600;               // 51200
  float* M2 = M1 + 51200;                // 6400
  float* M3 = M2 + 6400;                 // 800
  float* M4 = M3 + 800;                  // 400
  float* STATS = M4 + 400;               // 256
  float* NBUF  = STATS + 256;            // pad to 8
  float* fh    = NBUF + 8;               // f16 area, 16B aligned
  _Float16* F0 = (_Float16*)fh;
  _Float16* F1 = F0 + NBP;
  _Float16* F2 = F1 + NBP;
  _Float16* WT = F2 + NBP;               // repacked weights, <= 442368 halves

  auto zeroF = [&](_Float16* p, size_t halves) {
    int n4 = (int)(halves / 8);
    int gb = (n4 + 255) / 256; if (gb > 4096) gb = 4096;
    zero_f16_k<<<gb, 256, 0, stream>>>((v4i*)p, n4);
  };

  auto conv = [&](const _Float16* act16, const float* w, float* out,
                  int Ci, int Co, int Di, int Hi, int Wi,
                  int Do, int Ho, int Wo,
                  int kd, int kh, int kw, int sd, int sh, int sw,
                  int pd, int ph, int pw) {
    int taps = kd * kh * kw;
    int K32 = (Ci * taps + 31) & ~31;
    int lc = __builtin_ctz((unsigned)Ci);
    int tot = Co * K32;
    repack_w_k<<<(tot + 255) / 256, 256, 0, stream>>>(w, WT, K32, lc, taps, tot);
    int Mv = Do * Ho * Wo;
    dim3 gg((Mv + 255) / 256, Co / 16);
    int Hp = Hi + 2, Wp = Wi + 2;
    switch (lc) {
      case 4: conv3d_wmma<4><<<gg, 256, 0, stream>>>(act16, WT, out, taps,
                Hp, Wp, Do, Ho, Wo, sd, sh, sw, pd, ph, pw, kh, kw); break;
      case 5: conv3d_wmma<5><<<gg, 256, 0, stream>>>(act16, WT, out, taps,
                Hp, Wp, Do, Ho, Wo, sd, sh, sw, pd, ph, pw, kh, kw); break;
      case 6: conv3d_wmma<6><<<gg, 256, 0, stream>>>(act16, WT, out, taps,
                Hp, Wp, Do, Ho, Wo, sd, sh, sw, pd, ph, pw, kh, kw); break;
      default: conv3d_wmma<7><<<gg, 256, 0, stream>>>(act16, WT, out, taps,
                Hp, Wp, Do, Ho, Wo, sd, sh, sw, pd, ph, pw, kh, kw); break;
    }
  };
  auto bn = [&](const float* x, const float* res, const float* m,
                const float* g, const float* b, int C, int Mv, int relu,
                float* out, _Float16* out16, int H, int W) {
    bn_stats_k<<<C, 256, 0, stream>>>(x, m, STATS, Mv);
    dim3 gg((Mv + 255) / 256, C);
    bn_apply_k<<<gg, 256, 0, stream>>>(x, res, m, g, b, STATS, NBUF, out, out16,
                                       Mv, __builtin_ctz((unsigned)C), relu,
                                       H, W, H + 2, W + 2);
  };
  auto maskn = [&](const float* m, int Mv) {
    mask_sum_k<<<1, 256, 0, stream>>>(m, NBUF, Mv);
  };

  float* cur = B0v;  _Float16* cur16 = F0;
  float* sA  = B1v;  _Float16* sA16  = F1;
  float* sB  = B2v;

  auto block = [&](BP bp, const float* m, int C, int Di, int Hi, int Wi) {
    int Mv = Di * Hi * Wi;
    conv(cur16, bp.c1.w, sA, C, C, Di, Hi, Wi, Di, Hi, Wi, 3,3,3, 1,1,1, 1,1,1);
    bn(sA, nullptr, m, bp.c1.g, bp.c1.b, C, Mv, 1, sA, sA16, Hi, Wi);
    conv(sA16, bp.c2.w, sB, C, C, Di, Hi, Wi, Di, Hi, Wi, 3,3,3, 1,1,1, 1,1,1);
    bn(sB, cur, m, bp.c2.g, bp.c2.b, C, Mv, 1, cur, cur16, Hi, Wi);
  };
  auto down = [&](CP cp, const float* m, int Ci, int Co,
                  int Di, int Hi, int Wi, int Do, int Ho, int Wo) {
    conv(cur16, cp.w, sA, Ci, Co, Di, Hi, Wi, Do, Ho, Wo, 3,3,3, 2,2,2, 1,1,1);
    // re-zero both f16 buffers for the new stage geometry (fresh halos)
    size_t newHalves = (size_t)(Do + 2) * (Ho + 2) * (Wo + 2) * Co;
    zeroF(cur16, newHalves);
    zeroF(sA16, newHalves);
    bn(sA, nullptr, m, cp.g, cp.b, Co, Do * Ho * Wo, 1, sA, sA16, Ho, Wo);
    { float* t = cur; cur = sA; sA = t; }
    { _Float16* t = cur16; cur16 = sA16; sA16 = t; }
  };

  // ---- stage 0: mask + conv_input ------------------------------------------
  zeroF(F0, NBP); zeroF(F1, NBP); zeroF(F2, NBP);
  compute_mask_k<<<(409600 + 255) / 256, 256, 0, stream>>>(FEAT, M0, 16, 409600);
  maskn(M0, 409600);
  to_cl16_k<<<(6553600 + 255) / 256, 256, 0, stream>>>(FEAT, F2, 4, 409600,
                                                       160, 160, 162, 162);
  conv(F2, cin_.w, cur, 16, 16, 16,160,160, 16,160,160, 3,3,3, 1,1,1, 1,1,1);
  bn(cur, nullptr, M0, cin_.g, cin_.b, 16, 409600, 1, cur, cur16, 160, 160);
  block(b1[0], M0, 16, 16, 160, 160);
  block(b1[1], M0, 16, 16, 160, 160);

  // ---- stage 1: 8x80x80, 32ch ----------------------------------------------
  pool_mask_k<<<(51200 + 255) / 256, 256, 0, stream>>>(M0, M1, 16,160,160, 2,2,2);
  maskn(M1, 51200);
  down(c2d, M1, 16, 32, 16,160,160, 8,80,80);
  block(b2[0], M1, 32, 8, 80, 80);
  block(b2[1], M1, 32, 8, 80, 80);

  // ---- stage 2: 4x40x40, 64ch ----------------------------------------------
  pool_mask_k<<<(6400 + 255) / 256, 256, 0, stream>>>(M1, M2, 8,80,80, 2,2,2);
  maskn(M2, 6400);
  down(c3d, M2, 32, 64, 8,80,80, 4,40,40);
  block(b3[0], M2, 64, 4, 40, 40);
  block(b3[1], M2, 64, 4, 40, 40);

  // ---- stage 3: 2x20x20, 128ch ---------------------------------------------
  pool_mask_k<<<(800 + 255) / 256, 256, 0, stream>>>(M2, M3, 4,40,40, 2,2,2);
  maskn(M3, 800);
  down(c4d, M3, 64, 128, 4,40,40, 2,20,20);
  block(b4[0], M3, 128, 2, 20, 20);
  block(b4[1], M3, 128, 2, 20, 20);

  // ---- conv_extra: (3,1,1) stride (2,1,1) pad (1,0,0) -> [128,1,20,20] -----
  pool_mask_k<<<(400 + 255) / 256, 256, 0, stream>>>(M3, M4, 2,20,20, 2,1,1);
  maskn(M4, 400);
  conv(cur16, cex.w, sA, 128, 128, 2,20,20, 1,20,20, 3,1,1, 2,1,1, 1,0,0);
  bn(sA, nullptr, M4, cex.g, cex.b, 128, 400, 1, (float*)d_out, nullptr, 20, 20);
}
